// HardAttention_83717502534073
// MI455X (gfx1250) — matrix operand order repaired and checked by reading, not compiled, not run on hardware
//
#include <hip/hip_runtime.h>

// Gumbel-noise softmax attention for MI455X (gfx1250, wave32, WMMA).
// B=2, H=16, S=2048, D=128, fp32 in/out; matmuls run through
// v_wmma_f32_16x16x32_bf16 (bf16 A/B, f32 accum). Problem is memory-bound
// (~1.2 GB mandatory traffic -> ~50us at 23.3 TB/s), so bf16 WMMA keeps the
// matrix work far below the memory time, and the per-element Gumbel math is
// reduced to raw v_log_f32/v_exp_f32 (inputs provably in normal range):
//   exp(s + gumbel(u)) == exp2(s*log2e - log2(-ln2*log2(u+eps) + eps))

#define BDIM 2
#define HDIM 16
#define SDIM 2048
#define DDIM 128

typedef __bf16 bf16;
typedef bf16  v16bf __attribute__((ext_vector_type(16)));
typedef float v8f   __attribute__((ext_vector_type(8)));

#define QK_SCALE 0.08838834764831845f  /* 1/sqrt(128) */
#define EPSF     1e-10f
#define NEGINF   -1e9f
#define LN2F     0.6931471805599453f
#define LOG2EF   1.4426950408889634f
#define VTP      40                     /* padded V^T row stride (bf16), 80B -> 16B-aligned rows */

// Pack 16 consecutive floats (4 float4s) into a bf16 x16 fragment, with scale.
__device__ inline v16bf pack4(float4 f0, float4 f1, float4 f2, float4 f3, float s) {
  v16bf r;
  r[0]  = (bf16)(f0.x * s); r[1]  = (bf16)(f0.y * s);
  r[2]  = (bf16)(f0.z * s); r[3]  = (bf16)(f0.w * s);
  r[4]  = (bf16)(f1.x * s); r[5]  = (bf16)(f1.y * s);
  r[6]  = (bf16)(f1.z * s); r[7]  = (bf16)(f1.w * s);
  r[8]  = (bf16)(f2.x * s); r[9]  = (bf16)(f2.y * s);
  r[10] = (bf16)(f2.z * s); r[11] = (bf16)(f2.w * s);
  r[12] = (bf16)(f3.x * s); r[13] = (bf16)(f3.y * s);
  r[14] = (bf16)(f3.z * s); r[15] = (bf16)(f3.w * s);
  return r;
}

__global__ __launch_bounds__(64) void gumbel_attn_gfx1250(
    const float* __restrict__ Q, const float* __restrict__ K,
    const float* __restrict__ V, const float* __restrict__ U,
    const int* __restrict__ Mk, float* __restrict__ out,
    float* __restrict__ attn)
{
  extern __shared__ char smem[];
  float* eAll    = (float*)smem;                                   // 2 waves * 16*2048 f32 = 256 KB
  bf16*  vt      = (bf16*)(smem + 2 * 16 * SDIM * 4);              // V^T chunk: 128 x 40 bf16 = 10 KB
  float* rinvArr = (float*)(smem + 2 * 16 * SDIM * 4 + DDIM * VTP * 2); // 32 f32

  const int wave = threadIdx.x >> 5;
  const int lane = threadIdx.x & 31;
  const int half = lane >> 4;     // which 16-lane half of the wave
  const int col  = lane & 15;     // N index / A-matrix M index

  const int rowBlocks = SDIM / 32;
  const int hid = blockIdx.x / rowBlocks;   // flattened b*H + h
  const int rb  = blockIdx.x % rowBlocks;
  const int b   = hid / HDIM;
  const int q0  = rb * 32 + wave * 16;      // this wave's 16 query rows

  const float* Qh = Q + (size_t)hid * SDIM * DDIM;
  const float* Kh = K + (size_t)hid * SDIM * DDIM;
  const float* Vh = V + (size_t)hid * SDIM * DDIM;
  const float* Uh = U + (size_t)hid * SDIM * SDIM;
  const int*   Mb = Mk + (size_t)b  * SDIM * SDIM;
  float* outH  = out  + (size_t)hid * SDIM * DDIM;
  float* attnH = attn + (size_t)hid * SDIM * SDIM;

  float* eT = eAll + (size_t)wave * 16 * SDIM;   // this wave's 16 x 2048 exp-score tile

  // ---- Q tile as 4 bf16 A-fragments (16x32 each), 1/sqrt(d) folded in ----
  // contiguous k order used here (B matches): element i of lane (m=col, half) holds K = i + 16*half
  v16bf aq[4];
  {
    const float* qrow = Qh + (size_t)(q0 + col) * DDIM;
#pragma unroll
    for (int dc = 0; dc < 4; ++dc) {
      const float* p = qrow + dc * 32 + 16 * half;
      float4 f0 = *(const float4*)(p + 0);
      float4 f1 = *(const float4*)(p + 4);
      float4 f2 = *(const float4*)(p + 8);
      float4 f3 = *(const float4*)(p + 12);
      aq[dc] = pack4(f0, f1, f2, f3, QK_SCALE);
    }
  }

  // ---- Pass 1: scores + gumbel + exp, spill e to LDS, accumulate row sums ----
  float rs[8];
#pragma unroll
  for (int r = 0; r < 8; ++r) rs[r] = 0.0f;

  for (int nt = 0; nt < SDIM / 16; ++nt) {
    const int n0 = nt * 16;
    v8f acc = {0.f, 0.f, 0.f, 0.f, 0.f, 0.f, 0.f, 0.f};
    // B fragment (32x16 bf16), same contiguous k order as A: element e of lane (n=col, half) holds K = e + 16*half
    const float* krow = Kh + (size_t)(n0 + col) * DDIM + 16 * half;
#pragma unroll
    for (int dc = 0; dc < 4; ++dc) {
      float4 f0 = *(const float4*)(krow + dc * 32 + 0);
      float4 f1 = *(const float4*)(krow + dc * 32 + 4);
      float4 f2 = *(const float4*)(krow + dc * 32 + 8);
      float4 f3 = *(const float4*)(krow + dc * 32 + 12);
      v16bf bk = pack4(f0, f1, f2, f3, 1.0f);
      acc = __builtin_amdgcn_wmma_f32_16x16x32_bf16(
          false, aq[dc], false, bk, (short)0, acc, false, false);
    }
    // C layout: VGPR r, lane -> (M = r + 8*half, N = col)
#pragma unroll
    for (int r = 0; r < 8; ++r) {
      const int mrow = r + 8 * half;
      const size_t idx = (size_t)(q0 + mrow) * SDIM + n0 + col;
      float u  = Uh[idx];
      int   mk = Mb[idx];
      float sc = (mk == 0) ? NEGINF : acc[r];
      // exp(sc + gumbel) via raw hw log2/exp2; all intermediates are normal:
      //   l1 = log2(u+eps); t = -ln2*l1 + eps (= -ln(u+eps)+eps > 0)
      //   ev = exp2(sc*log2e - log2(t))  [outer ln2 cancels exp's log2e]
      float l1 = __builtin_amdgcn_logf(u + EPSF);
      float t  = __builtin_fmaf(-LN2F, l1, EPSF);
      float l2 = __builtin_amdgcn_logf(t);
      float ev = __builtin_amdgcn_exp2f(__builtin_fmaf(sc, LOG2EF, -l2));
      rs[r] += ev;
      eT[mrow * SDIM + n0 + col] = ev;
    }
  }

  // ---- row-sum reduction across the 16 lanes of each half ----
  float rinvv[8];
#pragma unroll
  for (int r = 0; r < 8; ++r) {
    float v = rs[r];
    v += __shfl_xor(v, 1, 16);
    v += __shfl_xor(v, 2, 16);
    v += __shfl_xor(v, 4, 16);
    v += __shfl_xor(v, 8, 16);
    rinvv[r] = 1.0f / v;
  }
  if (col == 0) {
#pragma unroll
    for (int r = 0; r < 8; ++r) rinvArr[wave * 16 + r + 8 * half] = rinvv[r];
  }

  // ---- single streaming write of normalized attn (float4 stores) ----
  for (int it = 0; it < (16 * SDIM / 4) / 32; ++it) {   // 256 iters
    const int flat = it * 128 + lane * 4;
    const int row  = flat >> 11;          // /2048
    const int c2   = flat & (SDIM - 1);
    float4 ev = *(const float4*)(eT + flat);
    float ri  = rinvArr[wave * 16 + row];
    float4 o4;
    o4.x = ev.x * ri; o4.y = ev.y * ri; o4.z = ev.z * ri; o4.w = ev.w * ri;
    *(float4*)(attnH + (size_t)(q0 + row) * SDIM + c2) = o4;
  }

  // ---- Pass 2: O = (e @ V) * rinv, V^T staged cooperatively in LDS ----
  v8f o[8];
#pragma unroll
  for (int dt = 0; dt < 8; ++dt)
    o[dt] = (v8f){0.f, 0.f, 0.f, 0.f, 0.f, 0.f, 0.f, 0.f};

  for (int nc = 0; nc < SDIM / 32; ++nc) {
    const int n0 = nc * 32;
    __syncthreads();
    // stage V[n0..n0+31][0..127] transposed as bf16: vt[d][n]
#pragma unroll
    for (int j = 0; j < 16; ++j) {
      const int fid  = j * 64 + threadIdx.x;  // 1024 float4s
      const int nrow = fid >> 5;
      const int dcol = (fid & 31) * 4;
      float4 v4 = *(const float4*)(Vh + (size_t)(n0 + nrow) * DDIM + dcol);
      vt[(dcol + 0) * VTP + nrow] = (bf16)v4.x;
      vt[(dcol + 1) * VTP + nrow] = (bf16)v4.y;
      vt[(dcol + 2) * VTP + nrow] = (bf16)v4.z;
      vt[(dcol + 3) * VTP + nrow] = (bf16)v4.w;
    }
    __syncthreads();
    // A-fragment (16x32) of e from LDS, in contiguous k order (k = 16*half + i), matching bv
    const float* erow = eT + (size_t)col * SDIM + n0 + 16 * half;
    float4 f0 = *(const float4*)(erow + 0);
    float4 f1 = *(const float4*)(erow + 4);
    float4 f2 = *(const float4*)(erow + 8);
    float4 f3 = *(const float4*)(erow + 12);
    v16bf ap = pack4(f0, f1, f2, f3, 1.0f);
#pragma unroll
    for (int dt = 0; dt < 8; ++dt) {
      // B element e of lane (n=col, half) = V[n0 + e + 16*half][dt*16 + col]
      const v16bf bv = *(const v16bf*)(vt + (dt * 16 + col) * VTP + 16 * half);
      o[dt] = __builtin_amdgcn_wmma_f32_16x16x32_bf16(
          false, ap, false, bv, (short)0, o[dt], false, false);
    }
  }

  // ---- epilogue: normalize rows and store out ----
#pragma unroll
  for (int dt = 0; dt < 8; ++dt) {
#pragma unroll
    for (int r = 0; r < 8; ++r) {
      const int mrow = r + 8 * half;
      outH[(size_t)(q0 + mrow) * DDIM + dt * 16 + col] = o[dt][r] * rinvv[r];
    }
  }
}

extern "C" void kernel_launch(void* const* d_in, const int* in_sizes, int n_in,
                              void* d_out, int out_size, void* d_ws, size_t ws_size,
                              hipStream_t stream) {
  (void)in_sizes; (void)n_in; (void)out_size; (void)d_ws; (void)ws_size;
  const float* Q = (const float*)d_in[0];
  const float* K = (const float*)d_in[1];
  const float* V = (const float*)d_in[2];
  const float* U = (const float*)d_in[3];
  const int*   M = (const int*)d_in[4];
  float* out  = (float*)d_out;
  float* attn = out + (size_t)BDIM * HDIM * SDIM * DDIM;  // tuple: (out, attn) flat

  const size_t shmem = (size_t)2 * 16 * SDIM * 4   // e tiles (2 waves)
                     + (size_t)DDIM * VTP * 2      // V^T staging
                     + 32 * 4;                     // rinv
  // gfx1250 WGP has 320 KB LDS; raise the dynamic-LDS cap.
  (void)hipFuncSetAttribute((const void*)gumbel_attn_gfx1250,
                            hipFuncAttributeMaxDynamicSharedMemorySize,
                            (int)shmem);
  dim3 grid(BDIM * HDIM * (SDIM / 32));  // 2048 workgroups of 64 threads (2 waves)
  gumbel_attn_gfx1250<<<grid, 64, shmem, stream>>>(Q, K, V, U, M, out, attn);
}
